// LogSSMLayer_62302795596611
// MI455X (gfx1250) — matrix operand order repaired and checked by reading, compiled, not run on hardware
//
#include <hip/hip_runtime.h>
#include <cstdint>
#include <cstddef>

// ---------------------------------------------------------------------------
// Problem constants (from the reference)
// ---------------------------------------------------------------------------
static constexpr int B  = 4;
static constexpr int S  = 2048;
static constexpr int D  = 1024;
static constexpr int H  = 8;
static constexpr int DH = 64;
static constexpr int HD = H * DH;          // 512
static constexpr int M  = B * S;           // 8192 rows for all GEMMs
static constexpr float SCALE = 0.125f;     // 1/sqrt(64)
static constexpr float EPS   = 1e-8f;
static constexpr float NEG_BIG = -1e30f;   // "-inf" that stays NaN-free under adds

// ---------------------------------------------------------------------------
// WMMA types (CDNA5 / gfx1250, wave32)
// ---------------------------------------------------------------------------
typedef __bf16        v16bf __attribute__((ext_vector_type(16)));
typedef float         v8f   __attribute__((ext_vector_type(8)));
typedef unsigned int  u32x4 __attribute__((ext_vector_type(4)));

union FragBF16 {
    u32x4 q[2];   // two 16B chunks
    v16bf v;      // 16 bf16 = 32B
};

// ---------------------------------------------------------------------------
// fp32 -> bf16 conversion
// ---------------------------------------------------------------------------
__global__ void f32_to_bf16_kernel(const float* __restrict__ in,
                                   __bf16* __restrict__ out, int n)
{
    int i = blockIdx.x * blockDim.x + threadIdx.x;
    if (i < n) out[i] = (__bf16)in[i];
}

// ---------------------------------------------------------------------------
// Y[M,N] = Xb[M,K] * Wb[N,K]^T  (bf16 in, fp32 out), v_wmma_f32_16x16x32_bf16.
//
// Register-blocked: one wave32 computes a 32x64 super-tile = 2 (M) x 4 (N)
// 16x16 WMMA tiles with 8 independent fp32 accumulators. Per 32-deep K-step:
// 2 A-frags + 4 B-frags = 12 b128 loads feed 8 independent WMMAs.
//
// Fragment layouts (ISA 7.12.2):
//  A (16x32 MxK): lane l -> row m0+(l&15); half-slots 0..7 = K (l>>4)*8+0..7,
//                 slots 8..15 = K (l>>4)*8+16..23  (two disjoint b128 runs)
//  B (32x16 KxN): lane l -> col n0+(l&15); same K slots as A (two disjoint b128 runs)
//  C/D: lane l, vgpr r -> row m0 + r + (l>>4)*8, col n0 + (l&15)
// ---------------------------------------------------------------------------
static constexpr int MT = 2;   // 16x16 tiles per wave in M  (32 rows)
static constexpr int NT = 4;   // 16x16 tiles per wave in N  (64 cols)

__global__ void gemm_bf16_wmma_kernel(const __bf16* __restrict__ Xb,  // M x K
                                      const __bf16* __restrict__ Wb,  // N x K
                                      float* __restrict__ Y,          // M x N
                                      int Mm, int N, int K)
{
    const int wave   = (blockIdx.x * blockDim.x + threadIdx.x) >> 5;
    const int lane   = threadIdx.x & 31;
    const int supN   = N >> 6;                 // 64-wide super-tiles in N
    const int sm     = wave / supN;            // 32-row super-tile index
    const int sn     = wave - sm * supN;
    if (sm * 32 >= Mm) return;

    const int l15 = lane & 15;
    const int lhi = lane >> 4;
    const int kaoff = lhi * 8;                 // A octet base
    const int kboff = lhi * 8;                 // B octet base

    size_t rowA[MT], rowB[NT];
#pragma unroll
    for (int mt = 0; mt < MT; ++mt)
        rowA[mt] = (size_t)(sm * 32 + mt * 16 + l15) * K;
#pragma unroll
    for (int nt = 0; nt < NT; ++nt)
        rowB[nt] = (size_t)(sn * 64 + nt * 16 + l15) * K;

    v8f acc[MT][NT];
#pragma unroll
    for (int mt = 0; mt < MT; ++mt)
#pragma unroll
        for (int nt = 0; nt < NT; ++nt)
            acc[mt][nt] = (v8f){};

    for (int kk = 0; kk < K; kk += 32) {
        FragBF16 fa[MT], fb[NT];
#pragma unroll
        for (int mt = 0; mt < MT; ++mt) {
            fa[mt].q[0] = *(const u32x4*)(Xb + rowA[mt] + kk + kaoff);
            fa[mt].q[1] = *(const u32x4*)(Xb + rowA[mt] + kk + kaoff + 16);
        }
#pragma unroll
        for (int nt = 0; nt < NT; ++nt) {
            fb[nt].q[0] = *(const u32x4*)(Wb + rowB[nt] + kk + kboff);
            fb[nt].q[1] = *(const u32x4*)(Wb + rowB[nt] + kk + kboff + 16);
        }
#pragma unroll
        for (int mt = 0; mt < MT; ++mt)
#pragma unroll
            for (int nt = 0; nt < NT; ++nt)
                acc[mt][nt] = __builtin_amdgcn_wmma_f32_16x16x32_bf16(
                                  false, fa[mt].v, false, fb[nt].v,
                                  (short)0, acc[mt][nt], false, false);
    }

#pragma unroll
    for (int mt = 0; mt < MT; ++mt) {
        const int rbase = sm * 32 + mt * 16 + lhi * 8;
#pragma unroll
        for (int nt = 0; nt < NT; ++nt) {
            const int col = sn * 64 + nt * 16 + l15;
#pragma unroll
            for (int r = 0; r < 8; ++r)
                Y[(size_t)(rbase + r) * N + col] = acc[mt][nt][r];
        }
    }
}

// ---------------------------------------------------------------------------
// Per-(B*S, h): b = SCALE * sum_d q*k ; a = -sum_d softplus(alpha + bias)
// ---------------------------------------------------------------------------
__global__ void compute_ab_kernel(const float* __restrict__ q,
                                  const float* __restrict__ k,
                                  const float* __restrict__ alpha,
                                  const float* __restrict__ alpha_b,
                                  float* __restrict__ bArr,
                                  float* __restrict__ aArr)
{
    int i = blockIdx.x * blockDim.x + threadIdx.x;   // over M*H
    if (i >= M * H) return;
    int h  = i & (H - 1);
    int bs = i >> 3;
    const float* qp = q     + (size_t)bs * HD + h * DH;
    const float* kp = k     + (size_t)bs * HD + h * DH;
    const float* ap = alpha + (size_t)bs * HD + h * DH;
    const float* bp = alpha_b + h * DH;

    float sb = 0.f, sa = 0.f;
#pragma unroll 4
    for (int d = 0; d < DH; ++d) {
        sb += qp[d] * kp[d];
        float x  = ap[d] + bp[d];
        float sp = (x > 20.f) ? x : __logf(1.f + __expf(x));   // stable softplus
        sa += sp;
    }
    bArr[i] = sb * SCALE;
    aArr[i] = -sa;
}

// ---------------------------------------------------------------------------
// Sequential log-space scan over S. One thread owns one (batch, d) column and
// carries all H=8 state chains in registers (ln/nm/sg). Per-step a/b scalars
// are staged through LDS in 256-step chunks. Output y stored as bf16 for the
// final WMMA projection. No cross-thread communication in the scan.
// ---------------------------------------------------------------------------
static constexpr int TCH = 256;   // time chunk staged in LDS

__global__ void scan_kernel(const float* __restrict__ A,   // (B,S,H)
                            const float* __restrict__ Bc,  // (B,S,H)
                            const float* __restrict__ V,   // (B,S,D)
                            __bf16* __restrict__ Ybf)      // (B,S,D)
{
    __shared__ float sA[TCH * H];   // 8 KB
    __shared__ float sB[TCH * H];   // 8 KB

    const int bb = blockIdx.y;
    const int d  = blockIdx.x * blockDim.x + threadIdx.x;

    float ln[H], nm[H], sg[H];
#pragma unroll
    for (int h = 0; h < H; ++h) { ln[h] = NEG_BIG; nm[h] = NEG_BIG; sg[h] = 1.f; }

    const size_t baseAB = (size_t)bb * S * H;
    const size_t baseV  = (size_t)bb * S * D + d;

    for (int t0 = 0; t0 < S; t0 += TCH) {
        __syncthreads();
        for (int j = threadIdx.x; j < TCH * H; j += blockDim.x) {
            sA[j] = A [baseAB + (size_t)t0 * H + j];
            sB[j] = Bc[baseAB + (size_t)t0 * H + j];
        }
        __syncthreads();

        for (int tt = 0; tt < TCH; ++tt) {
            const int t = t0 + tt;
            float v = V[baseV + (size_t)t * D];
            if (tt + 8 < TCH)   // hint the streaming v reads ahead
                __builtin_prefetch(&V[baseV + (size_t)(t + 8) * D], 0, 0);

            const float vs = (v >= 0.f) ? 1.f : -1.f;        // sign, 0 -> +1
            const float vl = __logf(fabsf(v) + EPS);

            float ysum = 0.f;
#pragma unroll
            for (int h = 0; h < H; ++h) {
                const float at = sA[tt * H + h];
                const float bt = sB[tt * H + h];

                // ln = lse(a + ln, b)
                const float x  = at + ln[h];
                const float mm = fmaxf(x, bt);
                const float lnn = mm + __logf(1.f + __expf(-fabsf(x - bt)));

                // signed log-sum state update
                const float dec = at + nm[h];
                const float nw  = bt + vl;
                const float m2  = fmaxf(dec, nw);
                const float ss  = sg[h] * __expf(dec - m2) + vs * __expf(nw - m2);
                const float nsg = (ss >= 0.f) ? 1.f : -1.f;  // sign, 0 -> +1
                const float nmn = m2 + __logf(fabsf(ss) + EPS);

                ln[h] = lnn; nm[h] = nmn; sg[h] = nsg;
                ysum += nsg * __expf(nmn - lnn);
            }
            Ybf[baseV + (size_t)t * D] = (__bf16)ysum;
        }
    }
}

// ---------------------------------------------------------------------------
// Host-side launch
// ---------------------------------------------------------------------------
static inline size_t align256(size_t x) { return (x + 255) & ~(size_t)255; }

extern "C" void kernel_launch(void* const* d_in, const int* /*in_sizes*/, int /*n_in*/,
                              void* d_out, int /*out_size*/, void* d_ws, size_t /*ws_size*/,
                              hipStream_t stream)
{
    const float* x   = (const float*)d_in[0];
    const float* Wq  = (const float*)d_in[1];
    const float* Wk  = (const float*)d_in[2];
    const float* Wv  = (const float*)d_in[3];
    const float* Waw = (const float*)d_in[4];
    const float* Wab = (const float*)d_in[5];
    const float* Wo  = (const float*)d_in[6];
    float* out = (float*)d_out;

    // ---- workspace layout -------------------------------------------------
    char* ws = (char*)d_ws;
    size_t off = 0;
    auto take = [&](size_t bytes) { char* p = ws + off; off += align256(bytes); return p; };

    __bf16* xbf  = (__bf16*)take((size_t)M * D  * sizeof(__bf16));
    __bf16* wqb  = (__bf16*)take((size_t)HD * D * sizeof(__bf16));
    __bf16* wkb  = (__bf16*)take((size_t)HD * D * sizeof(__bf16));
    __bf16* wab  = (__bf16*)take((size_t)HD * D * sizeof(__bf16));
    __bf16* wvb  = (__bf16*)take((size_t)D * D  * sizeof(__bf16));
    __bf16* wob  = (__bf16*)take((size_t)D * D  * sizeof(__bf16));
    float*  qf   = (float*) take((size_t)M * HD * sizeof(float));
    float*  kf   = (float*) take((size_t)M * HD * sizeof(float));
    float*  af   = (float*) take((size_t)M * HD * sizeof(float));
    float*  vf   = (float*) take((size_t)M * D  * sizeof(float));
    float*  bArr = (float*) take((size_t)M * H  * sizeof(float));
    float*  aArr = (float*) take((size_t)M * H  * sizeof(float));
    __bf16* ybf  = (__bf16*)take((size_t)M * D  * sizeof(__bf16));

    // ---- 1) fp32 -> bf16 conversions -------------------------------------
    auto conv = [&](const float* src, __bf16* dst, int n) {
        f32_to_bf16_kernel<<<(n + 255) / 256, 256, 0, stream>>>(src, dst, n);
    };
    conv(x,   xbf, M * D);
    conv(Wq,  wqb, HD * D);
    conv(Wk,  wkb, HD * D);
    conv(Waw, wab, HD * D);
    conv(Wv,  wvb, D * D);
    conv(Wo,  wob, D * D);

    // ---- 2) WMMA GEMMs: q, k, alpha, v ------------------------------------
    auto gemm = [&](const __bf16* Xb, const __bf16* Wb, float* Y, int N) {
        int waves  = (M / 32) * (N / 64);             // 32x64 super-tiles
        int blocks = waves / 8;                       // 8 waves (256 thr) / block
        gemm_bf16_wmma_kernel<<<blocks, 256, 0, stream>>>(Xb, Wb, Y, M, N, D);
    };
    gemm(xbf, wqb, qf, HD);
    gemm(xbf, wkb, kf, HD);
    gemm(xbf, wab, af, HD);
    gemm(xbf, wvb, vf, D);

    // ---- 3) per-(bs,h) reductions: b and a --------------------------------
    compute_ab_kernel<<<(M * H + 255) / 256, 256, 0, stream>>>(qf, kf, af, Wab, bArr, aArr);

    // ---- 4) sequential log-space scan -> y (bf16) -------------------------
    dim3 sgrid(D / 256, B);
    scan_kernel<<<sgrid, 256, 0, stream>>>(aArr, bArr, vf, ybf);

    // ---- 5) output projection: out = y @ Wo^T (WMMA, fp32 out) ------------
    gemm(ybf, wob, out, D);
}
